// MultiheadSelfAttention_75909251989532
// MI455X (gfx1250) — matrix operand order repaired and checked by reading, not compiled, not run on hardware
//
#include <hip/hip_runtime.h>

// ---------- types ----------
typedef __attribute__((ext_vector_type(16))) __bf16 v16bf;
typedef __attribute__((ext_vector_type(8)))  __bf16 v8bf;
typedef __attribute__((ext_vector_type(4)))  __bf16 v4bf;
typedef __attribute__((ext_vector_type(8)))  float  v8f;

// D = A(16x32) * B(32x16) + C, bf16 in / f32 accum
__device__ __forceinline__ v8f wmma_bf16(v16bf a, v16bf b, v8f c) {
  return __builtin_amdgcn_wmma_f32_16x16x32_bf16(
      /*neg_a=*/false, a, /*neg_b=*/false, b,
      /*c_mod=*/(short)0, c, /*reuse_a=*/false, /*reuse_b=*/false);
}

// A-matrix fragment 16x32 (ISA 7.12.2): lane holds row (lane&15);
// lo lanes K = {kBase+0..7, kBase+16..23}, hi lanes +8.
__device__ __forceinline__ v16bf load_frag_a(const __bf16* rowPtr, int kBase) {
  const int koff = kBase + (((threadIdx.x >> 4) & 1) << 3);
  v8bf lo = *(const v8bf*)(rowPtr + koff);
  v8bf hi = *(const v8bf*)(rowPtr + koff + 16);
  return __builtin_shufflevector(lo, hi, 0, 1, 2, 3, 4, 5, 6, 7,
                                 8, 9, 10, 11, 12, 13, 14, 15);
}

// B-matrix fragment 32x16: lane holds column (lane&15);
// lo lanes K = {kBase+0..7, kBase+16..23}, hi lanes +8.
// rowPtr points at the K-contiguous source row for this lane's column (B = srcRow^T).
__device__ __forceinline__ v16bf load_frag_b(const __bf16* rowPtr, int kBase) {
  const int koff = kBase + (((threadIdx.x >> 4) & 1) << 3);
  v8bf lo = *(const v8bf*)(rowPtr + koff);
  v8bf hi = *(const v8bf*)(rowPtr + koff + 16);
  return __builtin_shufflevector(lo, hi, 0, 1, 2, 3, 4, 5, 6, 7,
                                 8, 9, 10, 11, 12, 13, 14, 15);
}

// ---------- fp32 -> bf16 conversion ----------
__global__ __launch_bounds__(256) void cvt_f32_bf16(const float* __restrict__ s,
                                                    __bf16* __restrict__ d, int n) {
  int i = (blockIdx.x * 256 + threadIdx.x) * 4;
  if (i + 3 < n) {
    float4 v = *(const float4*)(s + i);
    v4bf o;
    o[0] = (__bf16)v.x; o[1] = (__bf16)v.y;
    o[2] = (__bf16)v.z; o[3] = (__bf16)v.w;
    *(v4bf*)(d + i) = o;
  }
}

// ---------- GEMM: C[M,N] = A[M,K] * W[N,K]^T ----------
// Block = 256 threads = 8 waves (4 x 2). Block tile 128x128, wave tile 32x64.
// MODE 0: f32 out row-major. MODE 1: bf16 out row-major.
// MODE 2: bf16 out scattered to Vt[((b*16+h)*64 + d)*2048 + l]  (r = b*2048+l, c = h*64+d)
template <int MODE>
__global__ __launch_bounds__(256) void gemm_xwT(const __bf16* __restrict__ A,
                                                const __bf16* __restrict__ W,
                                                void* __restrict__ Cout,
                                                int M, int N, int K) {
  const int lane15 = threadIdx.x & 15;
  const int rowOff = ((threadIdx.x >> 4) & 1) << 3;
  const int wave = threadIdx.x >> 5;
  const int mBase = blockIdx.x * 128 + (wave & 3) * 32;
  const int nBase = blockIdx.y * 128 + (wave >> 2) * 64;

  const __bf16* a0 = A + (size_t)(mBase + lane15) * K;
  const __bf16* a1 = a0 + (size_t)16 * K;
  const __bf16* w0 = W + (size_t)(nBase + lane15) * K;

  v8f acc[2][4];
#pragma unroll
  for (int i = 0; i < 2; ++i)
#pragma unroll
    for (int t = 0; t < 4; ++t) acc[i][t] = {};

  for (int k = 0; k < K; k += 32) {
    __builtin_prefetch(a0 + k + 128, 0, 1);  // global_prefetch_b8
    v16bf fa0 = load_frag_a(a0, k);
    v16bf fa1 = load_frag_a(a1, k);
#pragma unroll
    for (int t = 0; t < 4; ++t) {
      v16bf fb = load_frag_b(w0 + (size_t)(t * 16) * K, k);
      acc[0][t] = wmma_bf16(fa0, fb, acc[0][t]);
      acc[1][t] = wmma_bf16(fa1, fb, acc[1][t]);
    }
  }

#pragma unroll
  for (int i = 0; i < 2; ++i)
#pragma unroll
    for (int t = 0; t < 4; ++t)
#pragma unroll
      for (int e = 0; e < 8; ++e) {
        int r = mBase + i * 16 + rowOff + e;
        int c = nBase + t * 16 + lane15;
        float v = acc[i][t][e];
        if (MODE == 0) {
          ((float*)Cout)[(size_t)r * N + c] = v;
        } else if (MODE == 1) {
          ((__bf16*)Cout)[(size_t)r * N + c] = (__bf16)v;
        } else {
          int b_ = r >> 11, l = r & 2047;
          int h_ = c >> 6, d_ = c & 63;
          ((__bf16*)Cout)[(((size_t)(b_ * 16 + h_)) * 64 + d_) * 2048 + l] = (__bf16)v;
        }
      }
}

// ---------- Flash attention ----------
// Grid: (L/128, B*H). Block = 256 = 8 waves; each wave owns 16 query rows.
// Key tile = 64. Row-sum l is accumulated as a 5th WMMA accumulator (P x ones),
// eliminating the sum-side cross-lane shuffles entirely.
// Q,K row-major (B*L, 1024) with head h at cols h*64; Vt is (B*H, 64, 2048).
__global__ __launch_bounds__(256) void flash_attn(const __bf16* __restrict__ Q,
                                                  const __bf16* __restrict__ Km,
                                                  const __bf16* __restrict__ Vt,
                                                  const unsigned char* __restrict__ mask,
                                                  __bf16* __restrict__ O) {
  __shared__ __align__(16) __bf16 pbuf[8][16 * 64];
  const int lane15 = threadIdx.x & 15;
  const int rowOff = ((threadIdx.x >> 4) & 1) << 3;
  const int wave = threadIdx.x >> 5;
  const int bh = blockIdx.y;
  const int b = bh >> 4, h = bh & 15;
  const int qBase = blockIdx.x * 128 + wave * 16;

  const __bf16* qrow = Q + (size_t)(b * 2048 + qBase + lane15) * 1024 + h * 64;
  const v16bf aQ0 = load_frag_a(qrow, 0);
  const v16bf aQ1 = load_frag_a(qrow, 32);

  const __bf16* kbase = Km + (size_t)(b * 2048) * 1024 + h * 64;
  const __bf16* vrow0 = Vt + ((size_t)bh * 64 + lane15) * 2048;
  const unsigned char* mrow = mask + b * 2048;
  __bf16* pw = pbuf[wave];

  // All-ones B fragment: (P x ones)[i][n] = rowsum(P)[i] for every column n.
  v16bf ones;
#pragma unroll
  for (int e = 0; e < 16; ++e) ones[e] = (__bf16)1.0f;

  float m_i[8];
  v8f o0 = {}, o1 = {}, o2 = {}, o3 = {}, o4 = {};  // o4 = running row-sum l
#pragma unroll
  for (int i = 0; i < 8; ++i) m_i[i] = -1.0e30f;

  for (int kt = 0; kt < 2048; kt += 64) {
    // ---- scores: 4 x (16q x 16k), K-dim = Dh = 64 ----
    v8f s[4];
    float bias[4];
#pragma unroll
    for (int j = 0; j < 4; ++j) {
      const __bf16* kr = kbase + (size_t)(kt + j * 16 + lane15) * 1024;
      v8f sj = {};
      sj = wmma_bf16(aQ0, load_frag_b(kr, 0), sj);
      sj = wmma_bf16(aQ1, load_frag_b(kr, 32), sj);
      s[j] = sj;
      bias[j] = mrow[kt + j * 16 + lane15] ? -3.0e38f : 0.0f;
    }

    // ---- online softmax (max via half-wave butterfly; sum via WMMA below) ----
#pragma unroll
    for (int i = 0; i < 8; ++i) {
      float x0 = s[0][i] * 0.125f + bias[0];  // 1/sqrt(64)
      float x1 = s[1][i] * 0.125f + bias[1];
      float x2 = s[2][i] * 0.125f + bias[2];
      float x3 = s[3][i] * 0.125f + bias[3];
      float t = fmaxf(fmaxf(x0, x1), fmaxf(x2, x3));
#pragma unroll
      for (int off = 8; off > 0; off >>= 1) t = fmaxf(t, __shfl_xor(t, off, 16));
      float mn = fmaxf(m_i[i], t);
      float al = __expf(m_i[i] - mn);
      m_i[i] = mn;
      float p0 = __expf(x0 - mn);
      float p1 = __expf(x1 - mn);
      float p2 = __expf(x2 - mn);
      float p3 = __expf(x3 - mn);
      __bf16* prow = pw + (i + rowOff) * 64 + lane15;
      prow[0]  = (__bf16)p0;
      prow[16] = (__bf16)p1;
      prow[32] = (__bf16)p2;
      prow[48] = (__bf16)p3;
      o0[i] *= al; o1[i] *= al; o2[i] *= al; o3[i] *= al; o4[i] *= al;
    }

    // P (16x64) back as two A-fragments; DS ops are in-order within a wave.
    const v16bf aP0 = load_frag_a(pw + lane15 * 64, 0);
    const v16bf aP1 = load_frag_a(pw + lane15 * 64, 32);

    // ---- O += P * V  (4 dv tiles x 2 key chunks) ----
    o0 = wmma_bf16(aP0, load_frag_b(vrow0, kt), o0);
    o0 = wmma_bf16(aP1, load_frag_b(vrow0, kt + 32), o0);
    o1 = wmma_bf16(aP0, load_frag_b(vrow0 + (size_t)16 * 2048, kt), o1);
    o1 = wmma_bf16(aP1, load_frag_b(vrow0 + (size_t)16 * 2048, kt + 32), o1);
    o2 = wmma_bf16(aP0, load_frag_b(vrow0 + (size_t)32 * 2048, kt), o2);
    o2 = wmma_bf16(aP1, load_frag_b(vrow0 + (size_t)32 * 2048, kt + 32), o2);
    o3 = wmma_bf16(aP0, load_frag_b(vrow0 + (size_t)48 * 2048, kt), o3);
    o3 = wmma_bf16(aP1, load_frag_b(vrow0 + (size_t)48 * 2048, kt + 32), o3);
    // ---- l += rowsum(P) via ones-matrix WMMA ----
    o4 = wmma_bf16(aP0, ones, o4);
    o4 = wmma_bf16(aP1, ones, o4);
  }

#pragma unroll
  for (int i = 0; i < 8; ++i) {
    float l = o4[i];
    float inv = l > 0.0f ? 1.0f / l : 0.0f;
    size_t base = (size_t)(b * 2048 + qBase + rowOff + i) * 1024 + h * 64 + lane15;
    O[base + 0]  = (__bf16)(o0[i] * inv);
    O[base + 16] = (__bf16)(o1[i] * inv);
    O[base + 32] = (__bf16)(o2[i] * inv);
    O[base + 48] = (__bf16)(o3[i] * inv);
  }
}

// ---------- launch ----------
extern "C" void kernel_launch(void* const* d_in, const int* in_sizes, int n_in,
                              void* d_out, int out_size, void* d_ws, size_t ws_size,
                              hipStream_t stream) {
  const float* x = (const float*)d_in[0];
  const unsigned char* mask = (const unsigned char*)d_in[1];
  const float* Wq = (const float*)d_in[2];
  const float* Wk = (const float*)d_in[3];
  const float* Wv = (const float*)d_in[4];
  const float* Wo = (const float*)d_in[5];

  constexpr int Mrows = 4 * 2048;           // 8192
  constexpr int D = 1024;
  constexpr size_t NX = (size_t)Mrows * D;  // 8,388,608
  constexpr size_t NW = (size_t)D * D;      // 1,048,576

  __bf16* xb  = (__bf16*)d_ws;
  __bf16* wqb = xb + NX;
  __bf16* wkb = wqb + NW;
  __bf16* wvb = wkb + NW;
  __bf16* wob = wvb + NW;
  __bf16* qb  = wob + NW;
  __bf16* kb  = qb + NX;
  __bf16* vtb = kb + NX;
  __bf16* ob  = vtb + NX;   // total ~88 MiB of workspace

  cvt_f32_bf16<<<dim3((unsigned)(NX / 4 / 256)), 256, 0, stream>>>(x, xb, (int)NX);
  cvt_f32_bf16<<<dim3((unsigned)(NW / 4 / 256)), 256, 0, stream>>>(Wq, wqb, (int)NW);
  cvt_f32_bf16<<<dim3((unsigned)(NW / 4 / 256)), 256, 0, stream>>>(Wk, wkb, (int)NW);
  cvt_f32_bf16<<<dim3((unsigned)(NW / 4 / 256)), 256, 0, stream>>>(Wv, wvb, (int)NW);
  cvt_f32_bf16<<<dim3((unsigned)(NW / 4 / 256)), 256, 0, stream>>>(Wo, wob, (int)NW);

  dim3 gg(Mrows / 128, D / 128);  // 64 x 8
  gemm_xwT<1><<<gg, 256, 0, stream>>>(xb, wqb, qb, Mrows, D, D);
  gemm_xwT<1><<<gg, 256, 0, stream>>>(xb, wkb, kb, Mrows, D, D);
  gemm_xwT<2><<<gg, 256, 0, stream>>>(xb, wvb, vtb, Mrows, D, D);

  flash_attn<<<dim3(2048 / 128, 4 * 16), 256, 0, stream>>>(qb, kb, vtb, mask, ob);

  gemm_xwT<0><<<gg, 256, 0, stream>>>(ob, wob, (float*)d_out, Mrows, D, D);
}
